// HRAMPolicyNet_87488483819531
// MI455X (gfx1250) — matrix operand order repaired and checked by reading, not compiled, not run on hardware
//
#include <hip/hip_runtime.h>

typedef unsigned short u16;
typedef unsigned int u32;
typedef __attribute__((ext_vector_type(16))) __bf16 v16bf;
typedef __attribute__((ext_vector_type(8)))  float  v8f;

#define DEVINL __device__ __forceinline__

// ---------------- constants ----------------
constexpr int CB = 2048;      // batch
constexpr int CS = 115;       // state dim
constexpr int CE = 3072;      // key/query dim
constexpr int CD = 256;       // model dim
constexpr int CK = 10;        // top-k
constexpr int CA = 23;        // actions
constexpr int CN = 50000;     // num keys
constexpr int NTILES = CN / 16;     // 3125 (exact)
constexpr int NCHUNK = 16;          // column chunks for fused topk
constexpr int SCW = 8;              // waves per scores workgroup
constexpr int NLISTS = NCHUNK * SCW; // 128 partial lists per row

// ---------------- helpers ----------------
DEVINL u16 f2bf(float f) {
  u32 u = __float_as_uint(f);
  if ((u & 0x7fffffffu) > 0x7f800000u) return (u16)((u >> 16) | 0x0040u); // quiet NaN
  u32 r = u + 0x7fffu + ((u >> 16) & 1u);
  return (u16)(r >> 16);
}

union FragU { v16bf v; uint4 q[2]; };

// A fragment (16x32 bf16): lane holds row m0+(lane&15); element i<8 -> K=kb+i,
// i>=8 -> K=kb+8+i with kb=(lane>>4)*8  => two b128 loads at +kb and +kb+16.
DEVINL v16bf load_a_frag(const u16* Ap, long rowOff, int k0, int lane) {
  int kb = (lane >> 4) * 8;
  const u16* p = Ap + rowOff + k0 + kb;
  FragU f;
  f.q[0] = *(const uint4*)(p);
  f.q[1] = *(const uint4*)(p + 16);
  return f.v;
}
// B fragment (32x16 bf16): lane holds column n0+(lane&15); same K order as A:
// kb=(lane>>4)*8  => two b128 loads at +kb and +kb+16.
DEVINL v16bf load_b_frag(const u16* Bp, long rowOff, int k0, int lane) {
  int kb = (lane >> 4) * 8;
  const u16* p = Bp + rowOff + k0 + kb;
  FragU f;
  f.q[0] = *(const uint4*)(p);
  f.q[1] = *(const uint4*)(p + 16);
  return f.v;
}

DEVINL v8f wmma_bf16(v16bf a, v16bf b, v8f c) {
  return __builtin_amdgcn_wmma_f32_16x16x32_bf16(false, a, false, b, (short)0, c, false, false);
}

// sorted-descending top-10 insert; tie-break: lower index wins (jax top_k order)
DEVINL void topk_insert(float (&tv)[10], u32 (&ti)[10], float v, u32 id) {
  int cnt = 0;
#pragma unroll
  for (int t = 0; t < 10; ++t)
    cnt += ((tv[t] > v) || (tv[t] == v && ti[t] < id)) ? 1 : 0;
#pragma unroll
  for (int j = 9; j >= 1; --j) {
    if (j > cnt)       { tv[j] = tv[j-1]; ti[j] = ti[j-1]; }
    else if (j == cnt) { tv[j] = v;       ti[j] = id;      }
  }
  if (cnt == 0) { tv[0] = v; ti[0] = id; }
}
DEVINL bool beats(float v, u32 id, float tv9, u32 ti9) {
  return (v > tv9) || (v == tv9 && id < ti9);
}

// ---------------- elementwise f32 -> bf16 ----------------
__global__ void k_cvt_bf16(const float* __restrict__ in, u16* __restrict__ out, long n) {
  long i = (long)blockIdx.x * blockDim.x + threadIdx.x;
  if (i < n) out[i] = f2bf(in[i]);
}

// ---------------- key prep: bf16 copy + 1/max(||k||,1e-12) ----------------
__global__ void k_keys_prep(const float* __restrict__ keys, u16* __restrict__ kbf,
                            float* __restrict__ rkn) {
  int r = blockIdx.x, t = threadIdx.x;
  __shared__ float red[256];
  float ss = 0.f;
  long base = (long)r * CE;
#pragma unroll
  for (int i = 0; i < CE / 256; ++i) {
    float x = keys[base + i * 256 + t];
    kbf[base + i * 256 + t] = f2bf(x);
    ss += x * x;
  }
  red[t] = ss; __syncthreads();
  for (int s = 128; s >= 1; s >>= 1) { if (t < s) red[t] += red[t + s]; __syncthreads(); }
  if (t == 0) rkn[r] = 1.f / fmaxf(sqrtf(red[0]), 1e-12f);
}

// ---------------- query LN + nan_to_num + L2 norm -> bf16 ----------------
__global__ void k_ln_query(const float* __restrict__ q, const float* __restrict__ g,
                           const float* __restrict__ b, u16* __restrict__ out) {
  int r = blockIdx.x, t = threadIdx.x;
  __shared__ float red[256];
  float xv[CE / 256], yv[CE / 256];
  long base = (long)r * CE;
  float s = 0.f;
#pragma unroll
  for (int i = 0; i < CE / 256; ++i) { xv[i] = q[base + i * 256 + t]; s += xv[i]; }
  red[t] = s; __syncthreads();
  for (int st = 128; st >= 1; st >>= 1) { if (t < st) red[t] += red[t + st]; __syncthreads(); }
  float mu = red[0] / (float)CE; __syncthreads();
  s = 0.f;
#pragma unroll
  for (int i = 0; i < CE / 256; ++i) { float d = xv[i] - mu; s += d * d; }
  red[t] = s; __syncthreads();
  for (int st = 128; st >= 1; st >>= 1) { if (t < st) red[t] += red[t + st]; __syncthreads(); }
  float inv = rsqrtf(red[0] / (float)CE + 1e-5f); __syncthreads();
  s = 0.f;
#pragma unroll
  for (int i = 0; i < CE / 256; ++i) {
    float y = (xv[i] - mu) * inv * g[i * 256 + t] + b[i * 256 + t];
    if (__builtin_isnan(y)) y = 0.f;
    else if (__builtin_isinf(y)) y = (y > 0.f) ? 1.f : -1.f;
    yv[i] = y; s += y * y;
  }
  red[t] = s; __syncthreads();
  for (int st = 128; st >= 1; st >>= 1) { if (t < st) red[t] += red[t + st]; __syncthreads(); }
  float rn = 1.f / fmaxf(sqrtf(red[0]), 1e-12f); __syncthreads();
#pragma unroll
  for (int i = 0; i < CE / 256; ++i) out[base + i * 256 + t] = f2bf(yv[i] * rn);
}

// ---------------- generic row LN over 256 dims ----------------
__global__ void k_ln256(const float* __restrict__ in, const float* __restrict__ g,
                        const float* __restrict__ b, float* __restrict__ outF, int ldF,
                        u16* __restrict__ outB, int ldB) {
  int r = blockIdx.x, t = threadIdx.x;
  __shared__ float red[256];
  float x = in[(long)r * CD + t];
  red[t] = x; __syncthreads();
  for (int s = 128; s >= 1; s >>= 1) { if (t < s) red[t] += red[t + s]; __syncthreads(); }
  float mu = red[0] / (float)CD; __syncthreads();
  float dx = x - mu;
  red[t] = dx * dx; __syncthreads();
  for (int s = 128; s >= 1; s >>= 1) { if (t < s) red[t] += red[t + s]; __syncthreads(); }
  float y = dx * rsqrtf(red[0] / (float)CD + 1e-5f) * g[t] + b[t];
  if (outF) outF[(long)r * ldF + t] = y;
  if (outB) outB[(long)r * ldB + t] = f2bf(y);
}

// ---------------- tiled f32 GEMM: C = opt_relu(A @ W^T + bias) ----------------
__global__ void k_gemm_f32(const float* __restrict__ Ap, int lda,
                           const float* __restrict__ Wp, int ldw,
                           const float* __restrict__ bias,
                           float* __restrict__ Cp, int ldc,
                           int M, int N, int K, int relu) {
  __shared__ float As[16][17], Ws[16][17];
  int tx = threadIdx.x, ty = threadIdx.y;
  int row = blockIdx.x * 16 + ty;
  int wr  = blockIdx.y * 16 + ty;
  int col = blockIdx.y * 16 + tx;
  float acc = 0.f;
  for (int k0 = 0; k0 < K; k0 += 16) {
    As[ty][tx] = (row < M && k0 + tx < K) ? Ap[(long)row * lda + k0 + tx] : 0.f;
    Ws[ty][tx] = (wr  < N && k0 + tx < K) ? Wp[(long)wr  * ldw + k0 + tx] : 0.f;
    __syncthreads();
#pragma unroll
    for (int kk = 0; kk < 16; ++kk) acc += As[ty][kk] * Ws[tx][kk];
    __syncthreads();
  }
  if (row < M && col < N) {
    float v = acc + (bias ? bias[col] : 0.f);
    if (relu) v = fmaxf(v, 0.f);
    Cp[(long)row * ldc + col] = v;
  }
}

// ---------------- generic bf16 WMMA GEMM: C = opt_relu(A @ B^T + bias) -------
// 8 waves/block; wave = 16x32 tile; block = 64 rows x 64 cols.
// k-loops are software-pipelined: next fragments load while current WMMAs run.
__global__ void k_gemm_wmma(const u16* __restrict__ Ap, int lda,
                            const u16* __restrict__ Bp, int ldb,
                            const float* __restrict__ bias,
                            float* __restrict__ Cf, u16* __restrict__ Cbf, int ldc,
                            int M, int N, int K,
                            const int* __restrict__ gather, int relu) {
  int lane = threadIdx.x & 31, wave = threadIdx.x >> 5;
  int mtile = blockIdx.x * 4 + (wave & 3);
  int n0 = (blockIdx.y * 2 + (wave >> 2)) * 32;
  int m0 = mtile * 16;
  if (m0 >= M || n0 >= N) return;
  int arow = m0 + (lane & 15);
  int garow = gather ? gather[arow] : arow;
  long aoff  = (long)garow * lda;
  long boff0 = (long)(n0 + (lane & 15)) * ldb;
  long boff1 = boff0 + 16L * ldb;
  bool do1 = (n0 + 32) <= N;
  v8f c0 = {}; v8f c1 = {};
  if (do1) {
    v16bf a  = load_a_frag(Ap, aoff, 0, lane);
    v16bf b0 = load_b_frag(Bp, boff0, 0, lane);
    v16bf b1 = load_b_frag(Bp, boff1, 0, lane);
    for (int k0 = 32; k0 < K; k0 += 32) {
      v16bf an  = load_a_frag(Ap, aoff, k0, lane);
      v16bf b0n = load_b_frag(Bp, boff0, k0, lane);
      v16bf b1n = load_b_frag(Bp, boff1, k0, lane);
      c0 = wmma_bf16(a, b0, c0);
      c1 = wmma_bf16(a, b1, c1);
      a = an; b0 = b0n; b1 = b1n;
    }
    c0 = wmma_bf16(a, b0, c0);
    c1 = wmma_bf16(a, b1, c1);
  } else {
    v16bf a  = load_a_frag(Ap, aoff, 0, lane);
    v16bf b0 = load_b_frag(Bp, boff0, 0, lane);
    for (int k0 = 32; k0 < K; k0 += 32) {
      v16bf an  = load_a_frag(Ap, aoff, k0, lane);
      v16bf b0n = load_b_frag(Bp, boff0, k0, lane);
      c0 = wmma_bf16(a, b0, c0);
      a = an; b0 = b0n;
    }
    c0 = wmma_bf16(a, b0, c0);
  }
  int mbase = m0 + 8 * (lane >> 4);
  int nc0 = n0 + (lane & 15);
#pragma unroll
  for (int v = 0; v < 8; ++v) {
    float x = c0[v] + (bias ? bias[nc0] : 0.f);
    if (relu) x = fmaxf(x, 0.f);
    long o = (long)(mbase + v) * ldc + nc0;
    if (Cf)  Cf[o]  = x;
    if (Cbf) Cbf[o] = f2bf(x);
  }
  if (do1) {
    int nc1 = nc0 + 16;
#pragma unroll
    for (int v = 0; v < 8; ++v) {
      float x = c1[v] + (bias ? bias[nc1] : 0.f);
      if (relu) x = fmaxf(x, 0.f);
      long o = (long)(mbase + v) * ldc + nc1;
      if (Cf)  Cf[o]  = x;
      if (Cbf) Cbf[o] = f2bf(x);
    }
  }
}

// ---------------- fused scores (qn @ kraw^T * rknorm) + per-wave top-10 ------
__global__ void k_scores_topk(const u16* __restrict__ Q, const u16* __restrict__ Kb,
                              const float* __restrict__ rkn,
                              float* __restrict__ pv, u32* __restrict__ pi) {
  __shared__ float tile[SCW][16][17];
  int lane = threadIdx.x & 31, wave = threadIdx.x >> 5;
  int m0 = blockIdx.x * 16;
  long aoff = (long)(m0 + (lane & 15)) * CE;
  const int per = (NTILES + NCHUNK - 1) / NCHUNK;
  int t0 = blockIdx.y * per;
  int t1 = t0 + per; if (t1 > NTILES) t1 = NTILES;
  float tv[10]; u32 ti[10];
#pragma unroll
  for (int j = 0; j < 10; ++j) { tv[j] = -__builtin_inff(); ti[j] = 0xffffffffu; }

  for (int t = t0 + wave; t < t1; t += SCW) {
    int n0 = t * 16;
    long boff = (long)(n0 + (lane & 15)) * CE;
    if (t + SCW < t1)
      __builtin_prefetch(Kb + (long)((t + SCW) * 16 + (lane & 15)) * CE, 0, 0);
    // software-pipelined K loop over E=3072 (96 WMMAs)
    v8f c = {};
    v16bf a = load_a_frag(Q, aoff, 0, lane);
    v16bf b = load_b_frag(Kb, boff, 0, lane);
    for (int k0 = 32; k0 < CE; k0 += 32) {
      v16bf an = load_a_frag(Q, aoff, k0, lane);
      v16bf bn = load_b_frag(Kb, boff, k0, lane);
      c = wmma_bf16(a, b, c);
      a = an; b = bn;
    }
    c = wmma_bf16(a, b, c);

    float rn = rkn[n0 + (lane & 15)];
    int mm = 8 * (lane >> 4), nn = lane & 15;
#pragma unroll
    for (int v = 0; v < 8; ++v) tile[wave][mm + v][nn] = c[v] * rn;
    asm volatile("s_wait_dscnt 0" ::: "memory");
    if (lane < 16) {
#pragma unroll
      for (int n = 0; n < 16; ++n) {
        float v = tile[wave][lane][n];
        u32 id = (u32)(n0 + n);
        if (beats(v, id, tv[9], ti[9])) topk_insert(tv, ti, v, id);
      }
    }
    asm volatile("s_wait_dscnt 0" ::: "memory");
  }
  if (lane < 16) {
    int row = m0 + lane;
    int slot = blockIdx.y * SCW + wave;
    long base = ((long)row * NLISTS + slot) * 10;
#pragma unroll
    for (int j = 0; j < 10; ++j) { pv[base + j] = tv[j]; pi[base + j] = ti[j]; }
  }
}

// ---------------- merge 128 partial lists per row -> final top-10 indices ----
__global__ void k_merge_topk(const float* __restrict__ pv, const u32* __restrict__ pi,
                             int* __restrict__ outIdx) {
  int row = blockIdx.x * blockDim.x + threadIdx.x;
  if (row >= CB) return;
  float tv[10]; u32 ti[10];
#pragma unroll
  for (int j = 0; j < 10; ++j) { tv[j] = -__builtin_inff(); ti[j] = 0xffffffffu; }
  long base = (long)row * NLISTS * 10;
  for (int c = 0; c < NLISTS * 10; ++c) {
    float v = pv[base + c]; u32 id = pi[base + c];
    if (beats(v, id, tv[9], ti[9])) topk_insert(tv, ti, v, id);
  }
#pragma unroll
  for (int j = 0; j < 10; ++j) outIdx[row * 10 + j] = (int)ti[j];
}

// ---------------- 4-head attention over K=10, hd=64 ----------------
__global__ void k_attention(const float* __restrict__ qh, const float* __restrict__ kv,
                            float* __restrict__ ctx, float* __restrict__ attnw) {
  int b = blockIdx.x, t = threadIdx.x;
  int h = t >> 6, d = t & 63;
  __shared__ float red[256];
  __shared__ float logit[4][10];
  __shared__ float attn[4][10];
  float q = qh[(long)b * CD + h * 64 + d];
  const float* kvb = kv + (long)b * CK * 512;
  for (int k = 0; k < CK; ++k) {
    red[t] = q * kvb[k * 512 + h * 64 + d];
    __syncthreads();
    for (int s = 32; s >= 1; s >>= 1) { if (d < s) red[t] += red[t + s]; __syncthreads(); }
    if (d == 0) logit[h][k] = red[h * 64] * 0.125f;   // 1/sqrt(64)
    __syncthreads();
  }
  if (t < 4) {
    float mx = -__builtin_inff();
    for (int k = 0; k < CK; ++k) mx = fmaxf(mx, logit[t][k]);
    float s = 0.f;
    for (int k = 0; k < CK; ++k) { float e = __expf(logit[t][k] - mx); attn[t][k] = e; s += e; }
    float is = 1.f / s;
    for (int k = 0; k < CK; ++k) attn[t][k] *= is;
  }
  __syncthreads();
  float acc = 0.f;
  for (int k = 0; k < CK; ++k) acc += attn[h][k] * kvb[k * 512 + 256 + h * 64 + d];
  ctx[(long)b * CD + h * 64 + d] = acc;
  if (t < CK)
    attnw[(long)b * CK + t] = 0.25f * (attn[0][t] + attn[1][t] + attn[2][t] + attn[3][t]);
}

// ---------------- host launcher ----------------
static inline int ceil_div(int a, int b) { return (a + b - 1) / b; }

extern "C" void kernel_launch(void* const* d_in, const int* in_sizes, int n_in,
                              void* d_out, int out_size, void* d_ws, size_t ws_size,
                              hipStream_t stream) {
  const float* state  = (const float*)d_in[0];
  const float* re_w1  = (const float*)d_in[1];  const float* re_b1 = (const float*)d_in[2];
  const float* re_w2  = (const float*)d_in[3];  const float* re_b2 = (const float*)d_in[4];
  const float* re_w3  = (const float*)d_in[5];  const float* re_b3 = (const float*)d_in[6];
  const float* re_ln_g = (const float*)d_in[7]; const float* re_ln_b = (const float*)d_in[8];
  const float* de_w1  = (const float*)d_in[9];  const float* de_b1 = (const float*)d_in[10];
  const float* de_w2  = (const float*)d_in[11]; const float* de_b2 = (const float*)d_in[12];
  const float* de_ln_g = (const float*)d_in[13]; const float* de_ln_b = (const float*)d_in[14];
  const float* keys   = (const float*)d_in[15];
  const float* ad_w1  = (const float*)d_in[16]; const float* ad_b1 = (const float*)d_in[17];
  const float* ad_w2  = (const float*)d_in[18]; const float* ad_b2 = (const float*)d_in[19];
  const float* ad_ln_g = (const float*)d_in[20]; const float* ad_ln_b = (const float*)d_in[21];
  const float* ipw    = (const float*)d_in[22]; const float* ipb   = (const float*)d_in[23];
  const float* out_w  = (const float*)d_in[24]; const float* out_b = (const float*)d_in[25];
  const float* ah_w1  = (const float*)d_in[26]; const float* ah_b1 = (const float*)d_in[27];
  const float* ah_w2  = (const float*)d_in[28]; const float* ah_b2 = (const float*)d_in[29];
  const float* cr_w1  = (const float*)d_in[30]; const float* cr_b1 = (const float*)d_in[31];
  const float* cr_w2  = (const float*)d_in[32]; const float* cr_b2 = (const float*)d_in[33];
  const float* cr_w3  = (const float*)d_in[34]; const float* cr_b3 = (const float*)d_in[35];

  float* out      = (float*)d_out;
  float* out_log  = out;                        // 2048 x 23
  float* out_attn = out + (long)CB * CA;        // 2048 x 10
  float* out_val  = out + (long)CB * (CA + CK); // 2048 x 1

  // workspace bump allocator
  char* wsp = (char*)d_ws;
  size_t off = 0;
  auto alloc = [&](size_t bytes) -> void* {
    void* r = wsp + off; off = (off + bytes + 255) & ~(size_t)255; return r;
  };
  u16*  kraw   = (u16*)  alloc((size_t)CN * CE * 2);
  float* rkn   = (float*)alloc((size_t)CN * 4);
  u16*  qn     = (u16*)  alloc((size_t)CB * CE * 2);
  // phase block: encoder scratch reused later for top-k partial lists
  size_t phaseBytes = (size_t)CB * 512 * 4 * 2 + (size_t)CB * CE * 4; // h1+h2+qbuf
  char* phase  = (char*) alloc(phaseBytes);
  float* h1    = (float*)phase;
  float* h2    = h1 + (size_t)CB * 512;
  float* qbuf  = h2 + (size_t)CB * 512;
  float* pv    = (float*)phase;                           // reuses phase after qn ready
  u32*   pi    = (u32*)(phase + (size_t)CB * NLISTS * 10 * 4);
  int*  tki    = (int*) alloc((size_t)CB * CK * 4);
  u16*  aw1b   = (u16*) alloc((size_t)1024 * CE * 2);
  u16*  aw2b   = (u16*) alloc((size_t)CD * 1024 * 2);
  u16*  ipwb   = (u16*) alloc((size_t)3 * CD * CD * 2);
  u16*  a_bf   = (u16*) alloc((size_t)CB * CK * 1024 * 2);
  float* cpre  = (float*)alloc((size_t)CB * CK * CD * 4);
  u16*  cbf    = (u16*) alloc((size_t)CB * CK * CD * 2);
  float* kv    = (float*)alloc((size_t)CB * CK * 512 * 4);
  float* qhb   = (float*)alloc((size_t)CB * CD * 4);
  float* d1    = (float*)alloc((size_t)CB * CD * 4);
  float* apre  = (float*)alloc((size_t)CB * CD * 4);
  u16*  agbf   = (u16*) alloc((size_t)CB * CD * 2);
  float* comb  = (float*)alloc((size_t)CB * 512 * 4);
  float* ctx   = (float*)alloc((size_t)CB * CD * 4);
  float* hb    = (float*)alloc((size_t)CB * CD * 4);
  float* c1    = (float*)alloc((size_t)CB * CD * 4);
  float* c2    = (float*)alloc((size_t)CB * CD * 4);
  (void)ws_size; (void)n_in; (void)in_sizes; (void)out_size;

  dim3 blk2(16, 16);
  auto gemmF = [&](const float* A, int lda, const float* W, int ldw, const float* bias,
                   float* C, int ldc, int M, int N, int K, int relu) {
    dim3 grid(ceil_div(M, 16), ceil_div(N, 16));
    k_gemm_f32<<<grid, blk2, 0, stream>>>(A, lda, W, ldw, bias, C, ldc, M, N, K, relu);
  };
  auto gemmW = [&](const u16* A, int lda, const u16* B, int ldb, const float* bias,
                   float* Cf, u16* Cbf, int ldc, int M, int N, int K,
                   const int* gather, int relu) {
    dim3 grid(ceil_div(M, 64), ceil_div(N, 64));
    k_gemm_wmma<<<grid, 256, 0, stream>>>(A, lda, B, ldb, bias, Cf, Cbf, ldc,
                                          M, N, K, gather, relu);
  };

  // weight conversions to bf16
  {
    long n1 = (long)1024 * CE;
    k_cvt_bf16<<<(int)((n1 + 255) / 256), 256, 0, stream>>>(ad_w1, aw1b, n1);
    long n2 = (long)CD * 1024;
    k_cvt_bf16<<<(int)((n2 + 255) / 256), 256, 0, stream>>>(ad_w2, aw2b, n2);
    long n3 = (long)3 * CD * CD;
    k_cvt_bf16<<<(int)((n3 + 255) / 256), 256, 0, stream>>>(ipw, ipwb, n3);
  }
  // key prep
  k_keys_prep<<<CN, 256, 0, stream>>>(keys, kraw, rkn);

  // retrieval encoder -> query
  gemmF(state, CS, re_w1, CS, re_b1, h1, 512, CB, 512, CS, 1);
  gemmF(h1, 512, re_w2, 512, re_b2, h2, 512, CB, 512, 512, 1);
  gemmF(h2, 512, re_w3, 512, re_b3, qbuf, CE, CB, CE, 512, 0);
  k_ln_query<<<CB, 256, 0, stream>>>(qbuf, re_ln_g, re_ln_b, qn);

  // fused scores + top-k  (phase block now reused for partial lists)
  {
    dim3 grid(CB / 16, NCHUNK);
    k_scores_topk<<<grid, 256, 0, stream>>>(qn, kraw, rkn, pv, pi);
    k_merge_topk<<<ceil_div(CB, 64), 64, 0, stream>>>(pv, pi, tki);
  }

  // knowledge adapter on gathered keys
  gemmW(kraw, CE, aw1b, CE, ad_b1, nullptr, a_bf, 1024, CB * CK, 1024, CE, tki, 1);
  gemmW(a_bf, 1024, aw2b, 1024, ad_b2, cpre, nullptr, CD, CB * CK, CD, 1024, nullptr, 0);
  k_ln256<<<CB * CK, 256, 0, stream>>>(cpre, ad_ln_g, ad_ln_b, nullptr, 0, cbf, CD);

  // state decoder -> agent (into combined[:, :256] + bf16 copy)
  gemmF(state, CS, de_w1, CS, de_b1, d1, CD, CB, CD, CS, 1);
  gemmF(d1, CD, de_w2, CD, de_b2, apre, CD, CB, CD, CD, 0);
  k_ln256<<<CB, 256, 0, stream>>>(apre, de_ln_g, de_ln_b, comb, 512, agbf, CD);

  // q/k/v projections
  gemmW(cbf, CD, ipwb + (size_t)CD * CD, CD, ipb + CD, kv, nullptr, 512,
        CB * CK, 512, CD, nullptr, 0);
  gemmW(agbf, CD, ipwb, CD, ipb, qhb, nullptr, CD, CB, CD, CD, nullptr, 0);

  // attention (writes mean attn weights straight to output)
  k_attention<<<CB, 256, 0, stream>>>(qhb, kv, ctx, out_attn);

  // output projection into combined[:, 256:512]
  gemmF(ctx, CD, out_w, CD, out_b, comb + CD, 512, CB, CD, CD, 0);

  // action head
  gemmF(comb, 512, ah_w1, 512, ah_b1, hb, CD, CB, CD, 512, 1);
  gemmF(hb, CD, ah_w2, CD, ah_b2, out_log, CA, CB, CA, CD, 0);

  // critic
  gemmF(state, CS, cr_w1, CS, cr_b1, c1, CD, CB, CD, CS, 1);
  gemmF(c1, CD, cr_w2, CD, cr_b2, c2, CD, CB, CD, CD, 1);
  gemmF(c2, CD, cr_w3, CD, cr_b3, out_val, 1, CB, 1, CD, 0);
}
